// GraphRec_19937238188376
// MI455X (gfx1250) — compile-verified, matrix operand order checked by reading, not run on hardware
//
#include <hip/hip_runtime.h>

typedef __attribute__((ext_vector_type(16))) _Float16 v16h;
typedef __attribute__((ext_vector_type(8)))  float    v8f;
typedef int v4i __attribute__((vector_size(16)));

#if defined(__HIP_DEVICE_COMPILE__) && defined(__gfx1250__)
#if __has_builtin(__builtin_amdgcn_global_load_async_to_lds_b128) && \
    __has_builtin(__builtin_amdgcn_s_wait_asynccnt)
#define HAS_ASYNC_LDS 1
#endif
#endif

// ---------------------------------------------------------------- params ----
struct Params {
  // user
  const float *u_gv_l1W,*u_gv_l1b,*u_gv_l2W,*u_gv_l2b;
  const float *u_w1W,*u_w1b,*u_w4W,*u_w4b,*u_w5W,*u_w5b;
  const float *u_attui_l1W,*u_attui_l1b,*u_attui_l2W,*u_attui_l2b;
  const float *u_aggI_W,*u_aggI_b;
  const float *u_atts1_l1W,*u_atts1_l1b,*u_atts1_l2W,*u_atts1_l2b;
  const float *u_aggs1_W,*u_aggs1_b;
  const float *u_atts2_l1W,*u_atts2_l1b,*u_atts2_l2W,*u_atts2_l2b;
  const float *u_aggs2_W,*u_aggs2_b;
  const float *u_cmb_l1W,*u_cmb_l1b,*u_cmb_l2W,*u_cmb_l2b,*u_cmb_l3W,*u_cmb_l3b;
  // item
  const float *i_gu_l1W,*i_gu_l1b,*i_gu_l2W,*i_gu_l2b;
  const float *i_gv_l1W,*i_gv_l1b,*i_gv_l2W,*i_gv_l2b;
  const float *i_w1W,*i_w1b;
  const float *i_attiu_l1W,*i_attiu_l1b,*i_attiu_l2W,*i_attiu_l2b;
  const float *i_aggu_W,*i_aggu_b;
  const float *i_attui_l1W,*i_attui_l1b,*i_attui_l2W,*i_attui_l2b;
  const float *i_aggI_W,*i_aggI_b;
  const float *i_cmb_l1W,*i_cmb_l1b,*i_cmb_l2W,*i_cmb_l2b,*i_cmb_l3W,*i_cmb_l3b;
  // pred
  const float *p_l1W,*p_l1b,*p_l2W,*p_l2b,*p_l3W,*p_l3b;
};

enum { S_UGV1=0,S_UGV2,S_UW1,S_UW4,S_UW5,S_UATTUI,S_UATTS1,S_UATTS2,S_UAGGS1,
       S_IGU1,S_IGU2,S_IGV1,S_IGV2,S_IW1,S_IATTIU,S_IATTUI, NSLOT };
#define SLOT(w,i) ((w) + (size_t)(i)*8192)
#define TILES 16   // (b,p) tiles per block in k_user_social

// --------------------------------------------------------- LDS staging ------
// f16 global -> LDS, 16B per op; async path uses ASYNCcnt-tracked DMA.
__device__ __forceinline__ void stage_w(_Float16* dst, const _Float16* src,
                                        int nhalf, int tid) {
#ifdef HAS_ASYNC_LDS
  for (int i = tid * 8; i < nhalf; i += 128 * 8)
    __builtin_amdgcn_global_load_async_to_lds_b128(
        (__attribute__((address_space(1))) v4i*)(src + i),
        (__attribute__((address_space(3))) v4i*)(dst + i), 0, 0);
#else
  const uint4* s = (const uint4*)src;
  uint4*       d = (uint4*)dst;
  for (int i = tid; i < (nhalf >> 3); i += 128) d[i] = s[i];
#endif
}
__device__ __forceinline__ void stage_fence() {
#ifdef HAS_ASYNC_LDS
  __builtin_amdgcn_s_wait_asynccnt(0);
#endif
  __syncthreads();
}

// ------------------------------------------------------------ wmma helpers --
// 16-bit A/B fragment per CDNA5 ISA: lane 0-15 -> index lane, K {k0..k0+7, k0+16..k0+23};
// lane 16-31 -> K shifted by 8.
__device__ __forceinline__ v16h frag16(const _Float16* base, int stride, int idx0,
                                       int k0, int lane) {
  const _Float16* p = base + (size_t)(idx0 + (lane & 15)) * stride
                           + k0 + ((lane >> 4) << 3);
  v16h f;
#pragma unroll
  for (int i = 0; i < 8; ++i) { f[i] = p[i]; f[i + 8] = p[16 + i]; }
  return f;
}

// C = act(A(64xK) @ W(64xK)^T + bias), A/W f16 in LDS, out f16 (stride ostride).
// ACT: 0 none, 1 leaky(0.2), 2 relu. 4 waves: wave w owns rows [16w,16w+16).
template <int KDIM, int ACT>
__device__ __forceinline__ void gemm64(const _Float16* A, const _Float16* W,
                                       _Float16* Out, int ostride,
                                       const float* bias, int tid) {
  const int lane = tid & 31, wave = tid >> 5;
  const int row0 = wave << 4;
  v8f acc[4] = {};
#pragma unroll
  for (int kt = 0; kt < KDIM / 32; ++kt) {
    v16h af = frag16(A, KDIM, row0, kt * 32, lane);
#pragma unroll
    for (int nt = 0; nt < 4; ++nt) {
      v16h bf = frag16(W, KDIM, nt * 16, kt * 32, lane);
      acc[nt] = __builtin_amdgcn_wmma_f32_16x16x32_f16(
          false, af, false, bf, (short)0, acc[nt], false, false);
    }
  }
  const int n = lane & 15, hi = (lane >> 4) << 3;
#pragma unroll
  for (int nt = 0; nt < 4; ++nt) {
    float bb = bias[nt * 16 + n];
#pragma unroll
    for (int r = 0; r < 8; ++r) {
      float v = acc[nt][r] + bb;
      if (ACT == 1) v = v > 0.f ? v : 0.2f * v;
      else if (ACT == 2) v = fmaxf(v, 0.f);
      Out[(size_t)(row0 + hi + r) * ostride + nt * 16 + n] = (_Float16)v;
    }
  }
}

// ----------------------------------------------------------- VALU helpers ---
// second half of concat: lin(mask*u)[j][n] = mask[j]*(u@W^T)[n] + b[n]
__device__ __forceinline__ void uwfill(const float* W, const float* bias,
                                       const float* u, float* uw,
                                       const float* mask, _Float16* sA, int tid) {
  if (tid < 64) {
    float a = 0.f; const float* w = W + tid * 64;
    for (int k = 0; k < 64; ++k) a += u[k] * w[k];
    uw[tid] = a;
  }
  __syncthreads();
  for (int e = tid; e < 64 * 64; e += 128) {
    int j = e >> 6, n = e & 63;
    sA[j * 128 + 64 + n] = (_Float16)(mask[j] * uw[n] + bias[n]);
  }
  __syncthreads();
}

// attention head (dim-1 linear) + exp*mask -> e[64]
__device__ __forceinline__ void att_exp(const _Float16* Bm, const float* w2,
                                        const float* b2, const float* mask,
                                        float* e, int tid) {
  if (tid < 64) {
    float a = b2[0];
    for (int k = 0; k < 64; ++k) a += (float)Bm[tid * 64 + k] * w2[k];
    e[tid] = (tid < 50) ? __expf(a) * mask[tid] : 0.f;
  }
  __syncthreads();
}

// out[n] = sum_j (e_j / (sum e + eps)) * X[j][n]
__device__ __forceinline__ void wsum(const float* e, const _Float16* X,
                                     float* out, int tid) {
  if (tid < 64) {
    float s = 0.f;
    for (int j = 0; j < 50; ++j) s += e[j];
    float inv = 1.f / (s + 1e-10f);
    float a = 0.f;
    for (int j = 0; j < 50; ++j) a += e[j] * (float)X[j * 64 + tid];
    out[tid] = a * inv;
  }
  __syncthreads();
}

__device__ __forceinline__ void lin_relu64(const float* W, const float* bias,
                                           const float* x, float* y, int tid) {
  if (tid < 64) {
    float a = bias[tid]; const float* w = W + tid * 64;
    for (int k = 0; k < 64; ++k) a += w[k] * x[k];
    y[tid] = fmaxf(a, 0.f);
  }
  __syncthreads();
}

// _combine(concat(hA*hB, hA, hB)) -> out[64] (global)
__device__ __forceinline__ void combine_block(const float* hA, const float* hB,
    const float* l1W, const float* l1b, const float* l2W, const float* l2b,
    const float* l3W, const float* l3b,
    float* sC, float* sH1, float* sH2, float* out, int tid) {
  if (tid < 64) { sC[tid] = hA[tid] * hB[tid]; sC[64 + tid] = hA[tid]; sC[128 + tid] = hB[tid]; }
  __syncthreads();
  if (tid < 128) {
    float a = l1b[tid]; const float* w = l1W + tid * 192;
    for (int k = 0; k < 192; ++k) a += w[k] * sC[k];
    sH1[tid] = fmaxf(a, 0.f);
  }
  __syncthreads();
  if (tid < 64) {
    float a = l2b[tid]; const float* w = l2W + tid * 128;
    for (int k = 0; k < 128; ++k) a += w[k] * sH1[k];
    sH2[tid] = fmaxf(a, 0.f);
  }
  __syncthreads();
  if (tid < 64) {
    float a = l3b[tid]; const float* w = l3W + tid * 64;
    for (int k = 0; k < 64; ++k) a += w[k] * sH2[k];
    out[tid] = a;
  }
}

// gather 50 rows of concat(embA[id], embB[rate]) -> sA 64x128 f16 (zero-pad)
__device__ __forceinline__ void gather_rows(const int* pad, const float* embA,
                                            const float* embB, _Float16* sA,
                                            int tid) {
  for (int e = tid; e < 64 * 128; e += 128) {
    int j = e >> 7, c = e & 127;
    float v = 0.f;
    if (j < 50) {
      if (c < 64) { int id = pad[j * 2];     v = embA[(size_t)id * 64 + c]; }
      else        { int rt = pad[j * 2 + 1]; v = embB[(size_t)rt * 64 + c - 64]; }
    }
    sA[e] = (_Float16)v;
  }
}

// ----------------------------------------------------------------- kernels --
__global__ __launch_bounds__(256) void k_prep(Params P, _Float16* wgt) {
  const float* srcs[NSLOT] = {
    P.u_gv_l1W, P.u_gv_l2W, P.u_w1W, P.u_w4W, P.u_w5W, P.u_attui_l1W,
    P.u_atts1_l1W, P.u_atts2_l1W, P.u_aggs1_W, P.i_gu_l1W, P.i_gu_l2W,
    P.i_gv_l1W, P.i_gv_l2W, P.i_w1W, P.i_attiu_l1W, P.i_attui_l1W };
  const int cnts[NSLOT] = {8192,4096,4096,4096,4096,8192,8192,8192,4096,
                           8192,4096,8192,4096,4096,8192,8192};
  int s = blockIdx.x;
  const float* src = srcs[s];
  int n = cnts[s];
  for (int i = threadIdx.x; i < n; i += 256)
    wgt[(size_t)s * 8192 + i] = (_Float16)src[i];
}

// fused social path; persistent weights, TILES (b,p) tiles per block.
__global__ __launch_bounds__(128)
void k_user_social(const int* uui_pad, const int* uu_pad, const float* user_emb,
                   const float* item_emb, const float* rate_emb,
                   const _Float16* wgt, Params P, _Float16* hsum) {
  __shared__ __align__(16) _Float16 sWgv1[64 * 128];
  __shared__ __align__(16) _Float16 sWat1[64 * 128];
  __shared__ __align__(16) _Float16 sWgv2[64 * 64];
  __shared__ __align__(16) _Float16 sWw4[64 * 64];
  __shared__ __align__(16) _Float16 sA[64 * 128];
  __shared__ __align__(16) _Float16 sB[64 * 64];
  __shared__ __align__(16) _Float16 sX[64 * 64];
  __shared__ float sU[64], sUW[64], sMsk[64], sAl[64];

  const int tid = threadIdx.x;
  stage_w(sWgv1, SLOT(wgt, S_UGV1), 8192, tid);
  stage_w(sWgv2, SLOT(wgt, S_UGV2), 4096, tid);
  stage_w(sWw4,  SLOT(wgt, S_UW4),  4096, tid);
  stage_w(sWat1, SLOT(wgt, S_UATTS1), 8192, tid);
  stage_fence();

  for (int t = 0; t < TILES; ++t) {
    const int bp = blockIdx.x * TILES + t;
    const int* padp = uui_pad + (size_t)bp * 100;
    const int uid_s = uu_pad[bp];

    gather_rows(padp, item_emb, rate_emb, sA, tid);
    if (tid < 64) {
      sMsk[tid] = (tid < 50 && padp[tid * 2] > 0) ? 1.f : 0.f;
      sU[tid] = user_emb[(size_t)uid_s * 64 + tid];
    }
    __syncthreads();

    gemm64<128, 1>(sA, sWgv1, sB, 64, P.u_gv_l1b, tid);   // g_v.l1 + leaky
    __syncthreads();
    gemm64<64, 0>(sB, sWgv2, sX, 64, P.u_gv_l2b, tid);    // x_ia_s
    __syncthreads();
    gemm64<64, 0>(sX, sWw4, sA, 128, P.u_w4b, tid);       // w4(x) cols 0-63
    __syncthreads();
    uwfill(P.u_w4W, P.u_w4b, sU, sUW, sMsk, sA, tid);     // w4(p_i_s) cols 64-127
    gemm64<128, 1>(sA, sWat1, sB, 64, P.u_atts1_l1b, tid);
    __syncthreads();
    att_exp(sB, P.u_atts1_l2W, P.u_atts1_l2b, sMsk, sAl, tid);

    if (tid < 64) {
      float s = 0.f;
      for (int j = 0; j < 50; ++j) s += sAl[j];
      float inv = 1.f / (s + 1e-10f);
      float a = 0.f;
      for (int j = 0; j < 50; ++j) a += sAl[j] * (float)sX[j * 64 + tid];
      hsum[(size_t)bp * 64 + tid] = (_Float16)(a * inv);
    }
    __syncthreads();
  }
}

// user model main: one block per b -> h[b,64]
__global__ __launch_bounds__(128)
void k_user_main(const int* uids, const int* ui_pad, const int* uu_pad,
                 const float* user_emb, const float* item_emb,
                 const float* rate_emb, const _Float16* wgt,
                 const _Float16* hsum, Params P, float* h_out) {
  __shared__ __align__(16) _Float16 sWgv1[64 * 128];
  __shared__ __align__(16) _Float16 sWat1[64 * 128];
  __shared__ __align__(16) _Float16 sWat2[64 * 128];
  __shared__ __align__(16) _Float16 sWgv2[64 * 64];
  __shared__ __align__(16) _Float16 sWw1[64 * 64];
  __shared__ __align__(16) _Float16 sWw5[64 * 64];
  __shared__ __align__(16) _Float16 sWag1[64 * 64];
  __shared__ __align__(16) _Float16 sA[64 * 128];
  __shared__ __align__(16) _Float16 sB[64 * 64];
  __shared__ __align__(16) _Float16 sX[64 * 64];
  __shared__ float sU[64], sUW[64], sMu[64], sMs[64], sAl[64], sV[64];
  __shared__ float sHiI[64], sHiS[64], sC[192], sH1[128], sH2[64];

  const int tid = threadIdx.x, b = blockIdx.x;
  const int* padp = ui_pad + (size_t)b * 100;

  stage_w(sWgv1, SLOT(wgt, S_UGV1), 8192, tid);
  stage_w(sWgv2, SLOT(wgt, S_UGV2), 4096, tid);
  stage_w(sWw1,  SLOT(wgt, S_UW1),  4096, tid);
  stage_w(sWat1, SLOT(wgt, S_UATTUI), 8192, tid);
  stage_w(sWag1, SLOT(wgt, S_UAGGS1), 4096, tid);
  stage_w(sWw5,  SLOT(wgt, S_UW5),  4096, tid);
  stage_w(sWat2, SLOT(wgt, S_UATTS2), 8192, tid);

  gather_rows(padp, item_emb, rate_emb, sA, tid);   // overlaps async weight DMA
  if (tid < 64) {
    sMu[tid] = (tid < 50 && padp[tid * 2] > 0) ? 1.f : 0.f;
    sMs[tid] = (tid < 50 && uu_pad[(size_t)b * 50 + tid] > 0) ? 1.f : 0.f;
    sU[tid] = user_emb[(size_t)uids[b] * 64 + tid];
  }
  stage_fence();

  // item-history attention -> h_iI
  gemm64<128, 1>(sA, sWgv1, sB, 64, P.u_gv_l1b, tid);
  __syncthreads();
  gemm64<64, 0>(sB, sWgv2, sX, 64, P.u_gv_l2b, tid);      // x_ia
  __syncthreads();
  gemm64<64, 0>(sX, sWw1, sA, 128, P.u_w1b, tid);
  __syncthreads();
  uwfill(P.u_w1W, P.u_w1b, sU, sUW, sMu, sA, tid);
  gemm64<128, 1>(sA, sWat1, sB, 64, P.u_attui_l1b, tid);
  __syncthreads();
  att_exp(sB, P.u_attui_l2W, P.u_attui_l2b, sMu, sAl, tid);
  wsum(sAl, sX, sV, tid);
  lin_relu64(P.u_aggI_W, P.u_aggI_b, sV, sHiI, tid);

  // social aggregation -> h_iS
  for (int e = tid; e < 64 * 64; e += 128) {
    int j = e >> 6;
    sB[e] = (j < 50) ? hsum[((size_t)b * 50 + j) * 64 + (e & 63)] : (_Float16)0.f;
  }
  __syncthreads();
  gemm64<64, 2>(sB, sWag1, sX, 64, P.u_aggs1_b, tid);     // h_oI
  __syncthreads();
  gemm64<64, 0>(sX, sWw5, sA, 128, P.u_w5b, tid);
  __syncthreads();
  uwfill(P.u_w5W, P.u_w5b, sU, sUW, sMu, sA, tid);
  gemm64<128, 1>(sA, sWat2, sB, 64, P.u_atts2_l1b, tid);
  __syncthreads();
  att_exp(sB, P.u_atts2_l2W, P.u_atts2_l2b, sMs, sAl, tid);
  wsum(sAl, sX, sV, tid);
  lin_relu64(P.u_aggs2_W, P.u_aggs2_b, sV, sHiS, tid);

  combine_block(sHiI, sHiS, P.u_cmb_l1W, P.u_cmb_l1b, P.u_cmb_l2W, P.u_cmb_l2b,
                P.u_cmb_l3W, P.u_cmb_l3b, sC, sH1, sH2,
                h_out + (size_t)b * 64, tid);
}

// item model: one block per b -> z[b,64]
__global__ __launch_bounds__(128)
void k_item(const int* iids, const int* uids, const int* iu_pad,
            const int* ui_pad, const float* user_emb, const float* item_emb,
            const float* rate_emb, const _Float16* wgt, Params P,
            float* z_out) {
  __shared__ __align__(16) _Float16 sWgu1[64 * 128];
  __shared__ __align__(16) _Float16 sWatiu[64 * 128];
  __shared__ __align__(16) _Float16 sWgv1[64 * 128];
  __shared__ __align__(16) _Float16 sWatui[64 * 128];
  __shared__ __align__(16) _Float16 sWgu2[64 * 64];
  __shared__ __align__(16) _Float16 sWw1[64 * 64];
  __shared__ __align__(16) _Float16 sWgv2[64 * 64];
  __shared__ __align__(16) _Float16 sA[64 * 128];
  __shared__ __align__(16) _Float16 sB[64 * 64];
  __shared__ __align__(16) _Float16 sX[64 * 64];
  __shared__ __align__(16) _Float16 sP[64 * 64];
  __shared__ float sU[64], sUW[64], sMsk[64], sAl[64], sV[64];
  __shared__ float sZ[64], sHiI[64], sC[192], sH1[128], sH2[64];

  const int tid = threadIdx.x, b = blockIdx.x;

  stage_w(sWgu1,  SLOT(wgt, S_IGU1), 8192, tid);
  stage_w(sWgu2,  SLOT(wgt, S_IGU2), 4096, tid);
  stage_w(sWw1,   SLOT(wgt, S_IW1),  4096, tid);
  stage_w(sWatiu, SLOT(wgt, S_IATTIU), 8192, tid);
  stage_w(sWgv1,  SLOT(wgt, S_IGV1), 8192, tid);
  stage_w(sWgv2,  SLOT(wgt, S_IGV2), 4096, tid);
  stage_w(sWatui, SLOT(wgt, S_IATTUI), 8192, tid);

  // ---- pass A: rating users of this item -> z_j
  const int* padi = iu_pad + (size_t)b * 100;
  gather_rows(padi, user_emb, rate_emb, sA, tid);   // overlaps async weight DMA
  if (tid < 64) {
    sMsk[tid] = (tid < 50 && padi[tid * 2] > 0) ? 1.f : 0.f;
    sU[tid] = item_emb[(size_t)iids[b] * 64 + tid];
  }
  stage_fence();
  for (int e = tid; e < 64 * 64; e += 128)           // keep p_t
    sP[e] = sA[(e >> 6) * 128 + (e & 63)];
  __syncthreads();

  gemm64<128, 1>(sA, sWgu1, sB, 64, P.i_gu_l1b, tid);
  __syncthreads();
  gemm64<64, 0>(sB, sWgu2, sX, 64, P.i_gu_l2b, tid);      // f_jt
  __syncthreads();
  gemm64<64, 0>(sP, sWw1, sA, 128, P.i_w1b, tid);         // w1(p_t)
  __syncthreads();
  uwfill(P.i_w1W, P.i_w1b, sU, sUW, sMsk, sA, tid);       // w1(q_j)
  gemm64<128, 1>(sA, sWatiu, sB, 64, P.i_attiu_l1b, tid);
  __syncthreads();
  att_exp(sB, P.i_attiu_l2W, P.i_attiu_l2b, sMsk, sAl, tid);
  gemm64<64, 0>(sX, sWw1, sP, 64, P.i_w1b, tid);          // w1(f_jt)
  __syncthreads();
  wsum(sAl, sP, sV, tid);
  lin_relu64(P.i_aggu_W, P.i_aggu_b, sV, sZ, tid);        // z_j

  // ---- pass B: user's item history with item-model params -> h_iI
  const int* padu = ui_pad + (size_t)b * 100;
  gather_rows(padu, item_emb, rate_emb, sA, tid);
  if (tid < 64) {
    sMsk[tid] = (tid < 50 && padu[tid * 2] > 0) ? 1.f : 0.f;
    sU[tid] = user_emb[(size_t)uids[b] * 64 + tid];
  }
  __syncthreads();
  gemm64<128, 1>(sA, sWgv1, sB, 64, P.i_gv_l1b, tid);
  __syncthreads();
  gemm64<64, 0>(sB, sWgv2, sX, 64, P.i_gv_l2b, tid);
  __syncthreads();
  gemm64<64, 0>(sX, sWw1, sA, 128, P.i_w1b, tid);
  __syncthreads();
  uwfill(P.i_w1W, P.i_w1b, sU, sUW, sMsk, sA, tid);
  gemm64<128, 1>(sA, sWatui, sB, 64, P.i_attui_l1b, tid);
  __syncthreads();
  att_exp(sB, P.i_attui_l2W, P.i_attui_l2b, sMsk, sAl, tid);
  wsum(sAl, sX, sV, tid);
  lin_relu64(P.i_aggI_W, P.i_aggI_b, sV, sHiI, tid);

  combine_block(sZ, sHiI, P.i_cmb_l1W, P.i_cmb_l1b, P.i_cmb_l2W, P.i_cmb_l2b,
                P.i_cmb_l3W, P.i_cmb_l3b, sC, sH1, sH2,
                z_out + (size_t)b * 64, tid);
}

__global__ __launch_bounds__(64)
void k_pred(const float* h, const float* z, Params P, float* out) {
  __shared__ float sC[192], sH1[64], sH2[16];
  const int tid = threadIdx.x, b = blockIdx.x;
  if (tid < 64) {
    float hv = h[(size_t)b * 64 + tid], zv = z[(size_t)b * 64 + tid];
    sC[tid] = hv * zv; sC[64 + tid] = hv; sC[128 + tid] = zv;
  }
  __syncthreads();
  if (tid < 64) {
    float a = P.p_l1b[tid]; const float* w = P.p_l1W + tid * 192;
    for (int k = 0; k < 192; ++k) a += w[k] * sC[k];
    sH1[tid] = fmaxf(a, 0.f);
  }
  __syncthreads();
  if (tid < 16) {
    float a = P.p_l2b[tid]; const float* w = P.p_l2W + tid * 64;
    for (int k = 0; k < 64; ++k) a += w[k] * sH1[k];
    sH2[tid] = fmaxf(a, 0.f);
  }
  __syncthreads();
  if (tid == 0) {
    float a = P.p_l3b[0];
    for (int k = 0; k < 16; ++k) a += P.p_l3W[k] * sH2[k];
    out[b] = a;
  }
}

// ------------------------------------------------------------------ launch --
extern "C" void kernel_launch(void* const* d_in, const int* in_sizes, int n_in,
                              void* d_out, int out_size, void* d_ws,
                              size_t ws_size, hipStream_t stream) {
  (void)n_in; (void)out_size; (void)ws_size;
  const bool sorted = (in_sizes[0] == 25600);  // JAX-sorted pytree order starts with i_user_pad
  const int *uids, *iids, *ui_pad, *uu_pad, *uui_pad, *iu_pad;
  const float *user_emb, *item_emb, *rate_emb;
  Params P;
  auto F = [&](int i) { return (const float*)d_in[i]; };
  auto I = [&](int i) { return (const int*)d_in[i]; };

  if (!sorted) {  // dict insertion order
    uids = I(0); iids = I(1); ui_pad = I(2); uu_pad = I(3); uui_pad = I(4);
    iu_pad = I(5); user_emb = F(6); item_emb = F(7); rate_emb = F(8);
    int i = 9;
    P.u_gv_l1W=F(i++);P.u_gv_l1b=F(i++);P.u_gv_l2W=F(i++);P.u_gv_l2b=F(i++);
    P.u_w1W=F(i++);P.u_w1b=F(i++);P.u_w4W=F(i++);P.u_w4b=F(i++);
    P.u_w5W=F(i++);P.u_w5b=F(i++);
    P.u_attui_l1W=F(i++);P.u_attui_l1b=F(i++);P.u_attui_l2W=F(i++);P.u_attui_l2b=F(i++);
    P.u_aggI_W=F(i++);P.u_aggI_b=F(i++);
    P.u_atts1_l1W=F(i++);P.u_atts1_l1b=F(i++);P.u_atts1_l2W=F(i++);P.u_atts1_l2b=F(i++);
    P.u_aggs1_W=F(i++);P.u_aggs1_b=F(i++);
    P.u_atts2_l1W=F(i++);P.u_atts2_l1b=F(i++);P.u_atts2_l2W=F(i++);P.u_atts2_l2b=F(i++);
    P.u_aggs2_W=F(i++);P.u_aggs2_b=F(i++);
    P.u_cmb_l1W=F(i++);P.u_cmb_l1b=F(i++);P.u_cmb_l2W=F(i++);P.u_cmb_l2b=F(i++);
    P.u_cmb_l3W=F(i++);P.u_cmb_l3b=F(i++);
    P.i_gu_l1W=F(i++);P.i_gu_l1b=F(i++);P.i_gu_l2W=F(i++);P.i_gu_l2b=F(i++);
    P.i_gv_l1W=F(i++);P.i_gv_l1b=F(i++);P.i_gv_l2W=F(i++);P.i_gv_l2b=F(i++);
    P.i_w1W=F(i++);P.i_w1b=F(i++);
    P.i_attiu_l1W=F(i++);P.i_attiu_l1b=F(i++);P.i_attiu_l2W=F(i++);P.i_attiu_l2b=F(i++);
    P.i_aggu_W=F(i++);P.i_aggu_b=F(i++);
    P.i_attui_l1W=F(i++);P.i_attui_l1b=F(i++);P.i_attui_l2W=F(i++);P.i_attui_l2b=F(i++);
    P.i_aggI_W=F(i++);P.i_aggI_b=F(i++);
    P.i_cmb_l1W=F(i++);P.i_cmb_l1b=F(i++);P.i_cmb_l2W=F(i++);P.i_cmb_l2b=F(i++);
    P.i_cmb_l3W=F(i++);P.i_cmb_l3b=F(i++);
    P.p_l1W=F(i++);P.p_l1b=F(i++);P.p_l2W=F(i++);P.p_l2b=F(i++);
    P.p_l3W=F(i++);P.p_l3b=F(i++);
  } else {  // alphabetically-sorted pytree order
    iu_pad = I(0); iids = I(1); item_emb = F(2);
    int i = 3;
    P.i_aggI_W=F(i++);P.i_aggI_b=F(i++);
    P.i_aggu_W=F(i++);P.i_aggu_b=F(i++);
    P.i_attiu_l1W=F(i++);P.i_attiu_l1b=F(i++);P.i_attiu_l2W=F(i++);P.i_attiu_l2b=F(i++);
    P.i_attui_l1W=F(i++);P.i_attui_l1b=F(i++);P.i_attui_l2W=F(i++);P.i_attui_l2b=F(i++);
    P.i_cmb_l1W=F(i++);P.i_cmb_l1b=F(i++);P.i_cmb_l2W=F(i++);P.i_cmb_l2b=F(i++);
    P.i_cmb_l3W=F(i++);P.i_cmb_l3b=F(i++);
    P.i_gu_l1W=F(i++);P.i_gu_l1b=F(i++);P.i_gu_l2W=F(i++);P.i_gu_l2b=F(i++);
    P.i_gv_l1W=F(i++);P.i_gv_l1b=F(i++);P.i_gv_l2W=F(i++);P.i_gv_l2b=F(i++);
    P.i_w1W=F(i++);P.i_w1b=F(i++);
    P.p_l1W=F(i++);P.p_l1b=F(i++);P.p_l2W=F(i++);P.p_l2b=F(i++);
    P.p_l3W=F(i++);P.p_l3b=F(i++);
    P.u_aggI_W=F(i++);P.u_aggI_b=F(i++);
    P.u_aggs1_W=F(i++);P.u_aggs1_b=F(i++);
    P.u_aggs2_W=F(i++);P.u_aggs2_b=F(i++);
    P.u_atts1_l1W=F(i++);P.u_atts1_l1b=F(i++);P.u_atts1_l2W=F(i++);P.u_atts1_l2b=F(i++);
    P.u_atts2_l1W=F(i++);P.u_atts2_l1b=F(i++);P.u_atts2_l2W=F(i++);P.u_atts2_l2b=F(i++);
    P.u_attui_l1W=F(i++);P.u_attui_l1b=F(i++);P.u_attui_l2W=F(i++);P.u_attui_l2b=F(i++);
    P.u_cmb_l1W=F(i++);P.u_cmb_l1b=F(i++);P.u_cmb_l2W=F(i++);P.u_cmb_l2b=F(i++);
    P.u_cmb_l3W=F(i++);P.u_cmb_l3b=F(i++);
    P.u_gv_l1W=F(i++);P.u_gv_l1b=F(i++);P.u_gv_l2W=F(i++);P.u_gv_l2b=F(i++);
    P.u_w1W=F(i++);P.u_w1b=F(i++);
    P.u_w4W=F(i++);P.u_w4b=F(i++);
    P.u_w5W=F(i++);P.u_w5b=F(i++);
    rate_emb = F(i++); ui_pad = I(i++); uui_pad = I(i++); uu_pad = I(i++);
    uids = I(i++); user_emb = F(i++);
  }

  // workspace layout
  char* ws = (char*)d_ws;
  float*     h_buf = (float*)ws;                              // 256*64 f32
  float*     z_buf = (float*)(ws + 65536);                    // 256*64 f32
  _Float16*  hsum  = (_Float16*)(ws + 131072);                // 256*50*64 f16
  _Float16*  wgt   = (_Float16*)(ws + 131072 + 1638400);      // 16 slots * 8192 f16

  k_prep<<<NSLOT, 256, 0, stream>>>(P, wgt);
  k_user_social<<<(256 * 50) / TILES, 128, 0, stream>>>(uui_pad, uu_pad,
                                                        user_emb, item_emb,
                                                        rate_emb, wgt, P, hsum);
  k_user_main<<<256, 128, 0, stream>>>(uids, ui_pad, uu_pad, user_emb, item_emb,
                                       rate_emb, wgt, hsum, P, h_buf);
  k_item<<<256, 128, 0, stream>>>(iids, uids, iu_pad, ui_pad, user_emb,
                                  item_emb, rate_emb, wgt, P, z_buf);
  k_pred<<<256, 64, 0, stream>>>(h_buf, z_buf, P, (float*)d_out);
}
